// SpatiallyAwareGatedCrossAttention_72241349919413
// MI455X (gfx1250) — matrix operand order repaired and checked by reading, not compiled, not run on hardware
//
#include <hip/hip_runtime.h>
#include <math.h>

// ===========================================================================
// SpatiallyAwareGatedCrossAttention — MI455X (gfx1250) implementation.
//
// Strategy (reasoned from MI455X specs, compile-only):
//  * ~300 GFLOP of GEMM work vs ~0.5-0.7 GB of intermediate traffic.
//    HBM floor @23.3 TB/s ~30us; bf16 WMMA compute is far faster than that,
//    so: convert once to bf16, push every matmul (pointwise convs, gate/q/kv,
//    attention QK^T & PV, out-proj, FFN) through v_wmma_f32_16x16x32_bf16
//    (f32 accumulate), fuse all epilogues (bias/Swish/sigmoid/GELU/gated
//    residual) into the GEMM store so intermediates touch memory once.
//  * Attention is flash-style (online softmax) so the 8*16 (1024x1024) score
//    matrices never hit memory.
//  * Weight matrices are pre-transposed to (N,K) bf16 so WMMA B-fragments are
//    two contiguous 16B loads per lane (ISA 7.12.2 layouts).
//
// Input flattening assumption (setup_inputs() dict order, nested dicts in
// insertion order):
//   0 main_feat  1 side_feat
//   conv_main: 2 ln_g 3 ln_b 4 dw_w 5 dw_b 6 pw_w 7 pw_b
//   conv_side: 8 ln_g 9 ln_b 10 dw_w 11 dw_b 12 pw_w 13 pw_b
//   14 q_w 15 kv_w 16 out_w 17 out_b 18 gate_w 19 gate_b 20 rel_tab
//   21 ff1_w 22 ff1_b 23 ff2_w 24 ff2_b
//   25 nm_g 26 nm_b 27 ns_g 28 ns_b 29 npa_g 30 npa_b 31 npf_g 32 npf_b
// ===========================================================================

typedef __bf16 bf16;
typedef __attribute__((ext_vector_type(16))) __bf16 v16bf;
typedef __attribute__((ext_vector_type(8)))  float  v8f;
typedef __attribute__((ext_vector_type(4)))  unsigned int uint4v;
typedef __attribute__((ext_vector_type(4)))  float  float4v;

#define DIMC 1024
#define SEQC 1024
#define HC   16
#define HDC  64
#define FFC  4096
#define BC   8

__device__ inline v8f vzero8() {
  v8f z;
#pragma unroll
  for (int i = 0; i < 8; ++i) z[i] = 0.f;
  return z;
}

union FragBits { v16bf v; uint4v q[2]; };

// A-fragment (16x32 bf16, M x K) from row-major [rows x ld] bf16.
// ISA layout: lane m=lane&15, half=lane>>4; elems 0..7 = K half*8..half*8+7,
// elems 8..15 = K 16+half*8 .. 16+half*8+7.  (16B-aligned loads.)
__device__ inline v16bf load_frag_a(const bf16* __restrict__ base, int rowBase,
                                    int ld, int k0) {
  const int lane = threadIdx.x & 31;
  const int m = lane & 15, half = lane >> 4;
  const bf16* r = base + (size_t)(rowBase + m) * ld + k0;
  FragBits f;
  f.q[0] = *(const uint4v*)(r + half * 8);
  f.q[1] = *(const uint4v*)(r + 16 + half * 8);
  return f.v;
}

// B-fragment (32x16 bf16, K x N) from B^T stored row-major [N x ld(=K)] bf16.
// ISA layout: lane n=lane&15 is the column; elems 0..7 = K half*8..half*8+7,
// elems 8..15 = K 16+half*8 .. 16+half*8+7 (matches A-fragment order).
__device__ inline v16bf load_frag_bt(const bf16* __restrict__ baseT, int colBase,
                                     int ld, int k0) {
  const int lane = threadIdx.x & 31;
  const int n = lane & 15, half = lane >> 4;
  const bf16* r = baseT + (size_t)(colBase + n) * ld + k0;
  FragBits f;
  f.q[0] = *(const uint4v*)(r + half * 8);
  f.q[1] = *(const uint4v*)(r + 16 + half * 8);
  return f.v;
}

// ---------------------------------------------------------------------------
// Generic fused-epilogue bf16 GEMM:  C[M,N] = A[M,K] * Bt[N,K]^T  (+ epi)
// epi: 0 none(+bias)  1 swish(x+b)+res  2 sigmoid(x+b)  3 gelu(x+b)
//      4 res + (x+b)*gate              5 (x+b)+res
// ---------------------------------------------------------------------------
__device__ inline void store_frag(const v8f acc, int row0, int col0, int N,
                                  const float* __restrict__ bias,
                                  const float* __restrict__ res,
                                  const float* __restrict__ gatebuf,
                                  float* __restrict__ outF,
                                  bf16* __restrict__ outH, int epi) {
  const int lane = threadIdx.x & 31;
  const int n = lane & 15, ro = (lane >> 4) * 8;
  const int col = col0 + n;
  const float bv = bias ? bias[col] : 0.f;
#pragma unroll
  for (int r = 0; r < 8; ++r) {
    const int row = row0 + ro + r;
    const size_t idx = (size_t)row * N + col;
    float v = acc[r] + bv;
    if (epi == 1)      { v = v / (1.f + __expf(-v)); v += res[idx]; }
    else if (epi == 2) { v = 1.f / (1.f + __expf(-v)); }
    else if (epi == 3) { v = 0.5f * v * (1.f + erff(v * 0.70710678118f)); }
    else if (epi == 4) { v = res[idx] + v * gatebuf[idx]; }
    else if (epi == 5) { v = v + res[idx]; }
    if (outF) outF[idx] = v;
    if (outH) outH[idx] = (bf16)v;
  }
}

__global__ __launch_bounds__(256) void gemm_bf16_wmma(
    const bf16* __restrict__ A, const bf16* __restrict__ Bt,
    const float* __restrict__ bias, const float* __restrict__ res,
    const float* __restrict__ gatebuf, float* __restrict__ outF,
    bf16* __restrict__ outH, int M, int N, int K, int epi) {
  const int wave = threadIdx.x >> 5;
  const int wm = wave & 3, wn = wave >> 2;  // 4 x 2 wave grid -> 128 x 64 tile
  const int mBase = blockIdx.x * 128 + wm * 32;
  const int nBase = blockIdx.y * 64 + wn * 32;

  v8f acc00 = vzero8(), acc01 = vzero8(), acc10 = vzero8(), acc11 = vzero8();

  for (int k0 = 0; k0 < K; k0 += 32) {
    if (k0 + 32 < K) {  // L2 prefetch of next K-slab (global_prefetch_b8)
      __builtin_prefetch(A + (size_t)mBase * K + k0 + 32, 0, 1);
      __builtin_prefetch(Bt + (size_t)nBase * K + k0 + 32, 0, 1);
    }
    const v16bf a0 = load_frag_a(A, mBase, K, k0);
    const v16bf a1 = load_frag_a(A, mBase + 16, K, k0);
    const v16bf b0 = load_frag_bt(Bt, nBase, K, k0);
    const v16bf b1 = load_frag_bt(Bt, nBase + 16, K, k0);
    acc00 = __builtin_amdgcn_wmma_f32_16x16x32_bf16(false, a0, false, b0, (short)0, acc00, false, false);
    acc01 = __builtin_amdgcn_wmma_f32_16x16x32_bf16(false, a0, false, b1, (short)0, acc01, false, false);
    acc10 = __builtin_amdgcn_wmma_f32_16x16x32_bf16(false, a1, false, b0, (short)0, acc10, false, false);
    acc11 = __builtin_amdgcn_wmma_f32_16x16x32_bf16(false, a1, false, b1, (short)0, acc11, false, false);
  }
  store_frag(acc00, mBase,      nBase,      N, bias, res, gatebuf, outF, outH, epi);
  store_frag(acc01, mBase,      nBase + 16, N, bias, res, gatebuf, outF, outH, epi);
  store_frag(acc10, mBase + 16, nBase,      N, bias, res, gatebuf, outF, outH, epi);
  store_frag(acc11, mBase + 16, nBase + 16, N, bias, res, gatebuf, outF, outH, epi);
}

// ---------------------------------------------------------------------------
// ConvBlock stage 1: LayerNorm over L (per (b,d)) — stats pass.
// grid (B, D/64), block 256 = 64 d-lanes x 4 l-groups.
// ---------------------------------------------------------------------------
__global__ __launch_bounds__(256) void ln_over_L_stats(
    const float* __restrict__ x, float* __restrict__ meanO,
    float* __restrict__ rstdO, int B, int L, int D) {
  const int b = blockIdx.x;
  const int dl = threadIdx.x & 63;
  const int d = blockIdx.y * 64 + dl;
  const int lg = threadIdx.x >> 6;
  float s = 0.f, q = 0.f;
  for (int l = lg; l < L; l += 4) {
    const float v = x[((size_t)b * L + l) * D + d];
    s += v; q += v * v;
  }
  __shared__ float shS[4][64], shQ[4][64];
  shS[lg][dl] = s; shQ[lg][dl] = q;
  __syncthreads();
  if (threadIdx.x < 64) {
    const float ts = shS[0][dl] + shS[1][dl] + shS[2][dl] + shS[3][dl];
    const float tq = shQ[0][dl] + shQ[1][dl] + shQ[2][dl] + shQ[3][dl];
    const float mean = ts / (float)L;
    const float var = tq / (float)L - mean * mean;
    meanO[(size_t)b * D + d] = mean;
    rstdO[(size_t)b * D + d] = rsqrtf(var + 1e-5f);
  }
}

// ---------------------------------------------------------------------------
// ConvBlock stage 2: normalize-on-the-fly (gamma/beta indexed by L!) +
// depthwise conv (K=7, zero pad) -> bf16 output already in (B,L,D) layout
// (i.e. the transpose back is free) so the pointwise conv is a plain GEMM.
// ---------------------------------------------------------------------------
__global__ __launch_bounds__(256) void ln_dwconv_to_bf16(
    const float* __restrict__ x, const float* __restrict__ lng,
    const float* __restrict__ lnb, const float* __restrict__ dww,
    const float* __restrict__ dwb, const float* __restrict__ meanB,
    const float* __restrict__ rstdB, bf16* __restrict__ out, int B, int L,
    int D) {
  const size_t idx = (size_t)blockIdx.x * blockDim.x + threadIdx.x;
  const int d = (int)(idx & (D - 1));
  const int l = (int)((idx >> 10) & (L - 1));
  const int b = (int)(idx >> 20);
  const float mean = meanB[(size_t)b * D + d];
  const float rstd = rstdB[(size_t)b * D + d];
  float acc = dwb[d];
#pragma unroll
  for (int t = 0; t < 7; ++t) {
    const int ls = l + t - 3;
    if (ls >= 0 && ls < L) {
      float v = x[((size_t)b * L + ls) * D + d];
      v = (v - mean) * rstd * lng[ls] + lnb[ls];  // affine indexed by L pos
      acc += dww[d * 7 + t] * v;
    }
  }
  out[idx] = (bf16)acc;
}

// ---------------------------------------------------------------------------
// Row LayerNorm over D=1024 (one block per row), f32 in, f32/bf16 out.
// ---------------------------------------------------------------------------
__global__ __launch_bounds__(256) void layernorm_rows(
    const float* __restrict__ x, const float* __restrict__ g,
    const float* __restrict__ bt, float* __restrict__ outF,
    bf16* __restrict__ outH) {
  const int row = blockIdx.x, tid = threadIdx.x;
  const float* xr = x + (size_t)row * DIMC;
  const float4v v = ((const float4v*)xr)[tid];
  float s = v[0] + v[1] + v[2] + v[3];
  float q = v[0] * v[0] + v[1] * v[1] + v[2] * v[2] + v[3] * v[3];
#pragma unroll
  for (int mk = 1; mk < 32; mk <<= 1) {
    s += __shfl_xor(s, mk, 32);
    q += __shfl_xor(q, mk, 32);
  }
  __shared__ float ss[8], qq[8];
  if ((tid & 31) == 0) { ss[tid >> 5] = s; qq[tid >> 5] = q; }
  __syncthreads();
  s = 0.f; q = 0.f;
#pragma unroll
  for (int w = 0; w < 8; ++w) { s += ss[w]; q += qq[w]; }
  const float mean = s / (float)DIMC;
  const float rstd = rsqrtf(q / (float)DIMC - mean * mean + 1e-5f);
  const float4v gv = ((const float4v*)g)[tid];
  const float4v bv = ((const float4v*)bt)[tid];
  float4v o;
#pragma unroll
  for (int e = 0; e < 4; ++e) o[e] = (v[e] - mean) * rstd * gv[e] + bv[e];
  if (outF) ((float4v*)(outF + (size_t)row * DIMC))[tid] = o;
  if (outH) {
    bf16* oh = outH + (size_t)row * DIMC + tid * 4;
#pragma unroll
    for (int e = 0; e < 4; ++e) oh[e] = (bf16)o[e];
  }
}

// ---------------------------------------------------------------------------
// Weight converters (once per call; weights are runtime inputs).
// ---------------------------------------------------------------------------
__global__ void cvt_f32_to_bf16(const float* __restrict__ in,
                                bf16* __restrict__ out, size_t n) {
  const size_t i = (size_t)blockIdx.x * blockDim.x + threadIdx.x;
  if (i < n) out[i] = (bf16)in[i];
}
// in: K x N row-major (f32)  ->  out: N x K row-major (bf16)
__global__ void cvt_transpose_to_bf16(const float* __restrict__ in,
                                      bf16* __restrict__ out, int K, int N) {
  const size_t i = (size_t)blockIdx.x * blockDim.x + threadIdx.x;
  if (i >= (size_t)K * N) return;
  const int k = (int)(i % K);
  const int n = (int)(i / K);
  out[(size_t)n * K + k] = (bf16)in[(size_t)k * N + n];
}

// ---------------------------------------------------------------------------
// Flash attention with relative position bias.
// grid (B*H, S/128), block 256 (8 waves x 16 query rows).
// q: bf16 (B*S, 1024), kv: bf16 (B*S, 2048) [k cols 0..1023, v 1024..2047].
// ---------------------------------------------------------------------------
__global__ __launch_bounds__(256) void attention_flash_wmma(
    const bf16* __restrict__ qh, const bf16* __restrict__ kvh,
    const float* __restrict__ rel_tab, bf16* __restrict__ oh) {
  const int bh = blockIdx.x;
  const int b = bh / HC, h = bh % HC;
  const int tid = threadIdx.x, wave = tid >> 5, lane = tid & 31;
  const int i0 = blockIdx.y * 128 + wave * 16;
  const int n_lane = lane & 15, half = lane >> 4;

  __shared__ __align__(16) float rel_s[2 * SEQC - 1];
  __shared__ __align__(16) bf16 kT[32 * 72];       // K tile  [j=32][hd=64]+pad
  __shared__ __align__(16) bf16 vT[64 * 48];       // V^T tile [hd=64][j=32]+pad
  __shared__ __align__(16) bf16 pS[8][16 * 48];    // per-wave P restage 16x32+pad

  for (int t = tid; t < 2 * SEQC - 1; t += 256)
    rel_s[t] = rel_tab[(size_t)t * HC + h];

  // Q fragments for this wave's 16 rows (hd=64 -> two K-chunks).
  const bf16* qbase = qh + (size_t)b * SEQC * DIMC + h * HDC;
  const v16bf qa0 = load_frag_a(qbase, i0, DIMC, 0);
  const v16bf qa1 = load_frag_a(qbase, i0, DIMC, 32);

  float m_run[8], l_run[8];
#pragma unroll
  for (int r = 0; r < 8; ++r) { m_run[r] = -1e30f; l_run[r] = 0.f; }
  v8f accO[4];
#pragma unroll
  for (int c = 0; c < 4; ++c) accO[c] = vzero8();

  const bf16* kbase = kvh + (size_t)b * SEQC * 2048 + h * HDC;
  const bf16* vbase = kvh + (size_t)b * SEQC * 2048 + DIMC + h * HDC;
  const float scale = 0.125f;  // HD^-0.5

  for (int jb = 0; jb < SEQC; jb += 32) {
    __syncthreads();  // tiles from previous iteration fully consumed
    {                 // stage K tile and transposed V tile
      const int rowL = tid >> 3, cs = (tid & 7) * 8;
      const uint4v kq = *(const uint4v*)(kbase + (size_t)(jb + rowL) * 2048 + cs);
      *(uint4v*)(&kT[rowL * 72 + cs]) = kq;
      const uint4v vq = *(const uint4v*)(vbase + (size_t)(jb + rowL) * 2048 + cs);
      const bf16* vp = (const bf16*)&vq;
#pragma unroll
      for (int e = 0; e < 8; ++e) vT[(cs + e) * 48 + rowL] = vp[e];
    }
    __syncthreads();

    // S = Q K^T for two 16-wide j sub-tiles.
    v8f s0 = vzero8(), s1 = vzero8();
    {
      v16bf kb;
      kb = load_frag_bt(kT, 0, 72, 0);
      s0 = __builtin_amdgcn_wmma_f32_16x16x32_bf16(false, qa0, false, kb, (short)0, s0, false, false);
      kb = load_frag_bt(kT, 0, 72, 32);
      s0 = __builtin_amdgcn_wmma_f32_16x16x32_bf16(false, qa1, false, kb, (short)0, s0, false, false);
      kb = load_frag_bt(kT, 16, 72, 0);
      s1 = __builtin_amdgcn_wmma_f32_16x16x32_bf16(false, qa0, false, kb, (short)0, s1, false, false);
      kb = load_frag_bt(kT, 16, 72, 32);
      s1 = __builtin_amdgcn_wmma_f32_16x16x32_bf16(false, qa1, false, kb, (short)0, s1, false, false);
    }

    // Online softmax across these 32 columns (rows live across 16 lanes).
#pragma unroll
    for (int r = 0; r < 8; ++r) {
      const int i = i0 + r + half * 8;
      const int j0 = jb + n_lane, j1 = j0 + 16;
      float v0 = s0[r] * scale + rel_s[i - j0 + SEQC - 1];
      float v1 = s1[r] * scale + rel_s[i - j1 + SEQC - 1];
      float mx = fmaxf(v0, v1);
#pragma unroll
      for (int mk = 1; mk < 16; mk <<= 1) mx = fmaxf(mx, __shfl_xor(mx, mk, 32));
      const float mnew = fmaxf(m_run[r], mx);
      const float p0 = __expf(v0 - mnew), p1 = __expf(v1 - mnew);
      float rs = p0 + p1;
#pragma unroll
      for (int mk = 1; mk < 16; mk <<= 1) rs += __shfl_xor(rs, mk, 32);
      const float esc = __expf(m_run[r] - mnew);
      l_run[r] = l_run[r] * esc + rs;
      m_run[r] = mnew;
#pragma unroll
      for (int c = 0; c < 4; ++c) accO[c][r] *= esc;
      const int mrow = r + half * 8;
      pS[wave][mrow * 48 + n_lane] = (bf16)p0;
      pS[wave][mrow * 48 + 16 + n_lane] = (bf16)p1;
    }
    // same-wave LDS RAW: drain DScnt before reading P back in A layout
    __asm__ volatile("s_wait_dscnt 0x0" ::: "memory");

    const v16bf pa = load_frag_a(pS[wave], 0, 48, 0);
#pragma unroll
    for (int c = 0; c < 4; ++c) {
      const v16bf vb = load_frag_bt(vT, c * 16, 48, 0);
      accO[c] = __builtin_amdgcn_wmma_f32_16x16x32_bf16(false, pa, false, vb, (short)0, accO[c], false, false);
    }
  }

  // O /= l ; write head-interleaved bf16 output (B*S, 1024).
#pragma unroll
  for (int c = 0; c < 4; ++c)
#pragma unroll
    for (int r = 0; r < 8; ++r) {
      const int i = i0 + r + half * 8;
      const float ov = accO[c][r] / l_run[r];
      oh[((size_t)b * SEQC + i) * DIMC + h * HDC + c * 16 + n_lane] = (bf16)ov;
    }
}

// ===========================================================================
// Host-side orchestration
// ===========================================================================
extern "C" void kernel_launch(void* const* d_in, const int* in_sizes, int n_in,
                              void* d_out, int out_size, void* d_ws,
                              size_t ws_size, hipStream_t stream) {
  (void)in_sizes; (void)n_in; (void)out_size; (void)ws_size;
  const int B = BC, S = SEQC, D = DIMC, H = HC, FF = FFC;
  const size_t NT = (size_t)B * S;  // 8192 tokens

  const float* main_feat = (const float*)d_in[0];
  const float* side_feat = (const float*)d_in[1];
  const float* m_ln_g = (const float*)d_in[2];
  const float* m_ln_b = (const float*)d_in[3];
  const float* m_dw_w = (const float*)d_in[4];
  const float* m_dw_b = (const float*)d_in[5];
  const float* m_pw_w = (const float*)d_in[6];
  const float* m_pw_b = (const float*)d_in[7];
  const float* s_ln_g = (const float*)d_in[8];
  const float* s_ln_b = (const float*)d_in[9];
  const float* s_dw_w = (const float*)d_in[10];
  const float* s_dw_b = (const float*)d_in[11];
  const float* s_pw_w = (const float*)d_in[12];
  const float* s_pw_b = (const float*)d_in[13];
  const float* q_w    = (const float*)d_in[14];
  const float* kv_w   = (const float*)d_in[15];
  const float* out_w  = (const float*)d_in[16];
  const float* out_b  = (const float*)d_in[17];
  const float* gate_w = (const float*)d_in[18];
  const float* gate_b = (const float*)d_in[19];
  const float* rel_tab= (const float*)d_in[20];
  const float* ff1_w  = (const float*)d_in[21];
  const float* ff1_b  = (const float*)d_in[22];
  const float* ff2_w  = (const float*)d_in[23];
  const float* ff2_b  = (const float*)d_in[24];
  const float* nm_g   = (const float*)d_in[25];
  const float* nm_b   = (const float*)d_in[26];
  const float* ns_g   = (const float*)d_in[27];
  const float* ns_b   = (const float*)d_in[28];
  const float* npa_g  = (const float*)d_in[29];
  const float* npa_b  = (const float*)d_in[30];
  const float* npf_g  = (const float*)d_in[31];
  const float* npf_b  = (const float*)d_in[32];
  float* out = (float*)d_out;

  // ---- workspace bump allocator with lifetime-based aliasing --------------
  char* base = (char*)d_ws;
  size_t off = 0;
  auto take = [&](size_t bytes) -> char* {
    char* p = base + off;
    off = (off + bytes + 255) & ~(size_t)255;
    return p;
  };
  const size_t F32TOK = NT * D * sizeof(float);  // 32 MB
  const size_t H16TOK = NT * D * sizeof(bf16);   // 16 MB

  float* bufA  = (float*)take(F32TOK);   // mc -> preA -> preF
  float* bufBx = (float*)take(F32TOK);   // sc -> gate
  float* mws   = (float*)take(F32TOK);
  bf16* buf3   = (bf16*)take(H16TOK);    // a_main -> q_h
  bf16* buf4   = (bf16*)take(H16TOK);    // a_side -> o_h
  bf16* buf5   = (bf16*)take(H16TOK);    // mn_h -> mws_h
  bf16* sn_h   = (bf16*)take(H16TOK);
  bf16* kv_h   = (bf16*)take(NT * 2 * D * sizeof(bf16));   // 32 MB
  bf16* h_h    = (bf16*)take(NT * (size_t)FF * sizeof(bf16));  // 64 MB
  bf16* pwmT   = (bf16*)take((size_t)D * D * sizeof(bf16));
  bf16* pwsT   = (bf16*)take((size_t)D * D * sizeof(bf16));
  bf16* gateT  = (bf16*)take((size_t)D * D * sizeof(bf16));
  bf16* qT     = (bf16*)take((size_t)D * D * sizeof(bf16));
  bf16* kvT    = (bf16*)take((size_t)D * 2 * D * sizeof(bf16));
  bf16* outT   = (bf16*)take((size_t)D * D * sizeof(bf16));
  bf16* ff1T   = (bf16*)take((size_t)D * FF * sizeof(bf16));
  bf16* ff2T   = (bf16*)take((size_t)FF * D * sizeof(bf16));
  float* mean_m = (float*)take(NT * sizeof(float) / S * D);  // B*D floats
  float* rstd_m = (float*)take((size_t)B * D * sizeof(float));
  float* mean_s = (float*)take((size_t)B * D * sizeof(float));
  float* rstd_s = (float*)take((size_t)B * D * sizeof(float));

  const dim3 blk(256);
  auto g1 = [](size_t n) { return dim3((unsigned)((n + 255) / 256)); };

  // ---- weight conversion ---------------------------------------------------
  // pw_w is (out,in): already N x K for our B^T convention -> plain convert.
  cvt_f32_to_bf16<<<g1((size_t)D * D), blk, 0, stream>>>(m_pw_w, pwmT, (size_t)D * D);
  cvt_f32_to_bf16<<<g1((size_t)D * D), blk, 0, stream>>>(s_pw_w, pwsT, (size_t)D * D);
  // (K,N) weights -> transpose to (N,K).
  cvt_transpose_to_bf16<<<g1((size_t)D * D), blk, 0, stream>>>(gate_w, gateT, D, D);
  cvt_transpose_to_bf16<<<g1((size_t)D * D), blk, 0, stream>>>(q_w, qT, D, D);
  cvt_transpose_to_bf16<<<g1((size_t)D * 2 * D), blk, 0, stream>>>(kv_w, kvT, D, 2 * D);
  cvt_transpose_to_bf16<<<g1((size_t)D * D), blk, 0, stream>>>(out_w, outT, D, D);
  cvt_transpose_to_bf16<<<g1((size_t)D * FF), blk, 0, stream>>>(ff1_w, ff1T, D, FF);
  cvt_transpose_to_bf16<<<g1((size_t)FF * D), blk, 0, stream>>>(ff2_w, ff2T, FF, D);

  // ---- conv blocks ---------------------------------------------------------
  ln_over_L_stats<<<dim3(B, D / 64), blk, 0, stream>>>(main_feat, mean_m, rstd_m, B, S, D);
  ln_over_L_stats<<<dim3(B, D / 64), blk, 0, stream>>>(side_feat, mean_s, rstd_s, B, S, D);
  ln_dwconv_to_bf16<<<g1(NT * D), blk, 0, stream>>>(main_feat, m_ln_g, m_ln_b, m_dw_w,
                                                    m_dw_b, mean_m, rstd_m, buf3, B, S, D);
  ln_dwconv_to_bf16<<<g1(NT * D), blk, 0, stream>>>(side_feat, s_ln_g, s_ln_b, s_dw_w,
                                                    s_dw_b, mean_s, rstd_s, buf4, B, S, D);
  const dim3 gD(NT / 128, D / 64);
  // pointwise conv + bias + swish + residual  -> mc / sc (f32)
  gemm_bf16_wmma<<<gD, blk, 0, stream>>>(buf3, pwmT, m_pw_b, main_feat, nullptr,
                                         bufA, nullptr, (int)NT, D, D, 1);
  gemm_bf16_wmma<<<gD, blk, 0, stream>>>(buf4, pwsT, s_pw_b, side_feat, nullptr,
                                         bufBx, nullptr, (int)NT, D, D, 1);

  // ---- projections ---------------------------------------------------------
  layernorm_rows<<<dim3((unsigned)NT), blk, 0, stream>>>(bufA, nm_g, nm_b, nullptr, buf5);  // mn
  layernorm_rows<<<dim3((unsigned)NT), blk, 0, stream>>>(bufBx, ns_g, ns_b, nullptr, sn_h); // sn
  gemm_bf16_wmma<<<gD, blk, 0, stream>>>(buf5, gateT, gate_b, nullptr, nullptr,
                                         bufBx, nullptr, (int)NT, D, D, 2);  // gate (sigmoid)
  gemm_bf16_wmma<<<gD, blk, 0, stream>>>(buf5, qT, nullptr, nullptr, nullptr,
                                         nullptr, buf3, (int)NT, D, D, 0);   // q -> bf16
  gemm_bf16_wmma<<<dim3(NT / 128, 2 * D / 64), blk, 0, stream>>>(
      sn_h, kvT, nullptr, nullptr, nullptr, nullptr, kv_h, (int)NT, 2 * D, D, 0);  // kv

  // ---- attention -----------------------------------------------------------
  attention_flash_wmma<<<dim3(B * H, S / 128), blk, 0, stream>>>(buf3, kv_h, rel_tab, buf4);

  // ---- out proj + gated residual, LN, FFN, final LN ------------------------
  gemm_bf16_wmma<<<gD, blk, 0, stream>>>(buf4, outT, out_b, main_feat, bufBx,
                                         bufA, nullptr, (int)NT, D, D, 4);   // preA
  layernorm_rows<<<dim3((unsigned)NT), blk, 0, stream>>>(bufA, npa_g, npa_b, mws, buf5);  // mws
  gemm_bf16_wmma<<<dim3(NT / 128, FF / 64), blk, 0, stream>>>(
      buf5, ff1T, ff1_b, nullptr, nullptr, nullptr, h_h, (int)NT, FF, D, 3);  // gelu
  gemm_bf16_wmma<<<gD, blk, 0, stream>>>(h_h, ff2T, ff2_b, mws, nullptr,
                                         bufA, nullptr, (int)NT, D, FF, 5);  // preF
  layernorm_rows<<<dim3((unsigned)NT), blk, 0, stream>>>(bufA, npf_g, npf_b, out, nullptr);
}
